// PSAM_4380866642499
// MI455X (gfx1250) — matrix operand order repaired and checked by reading, not compiled, not run on hardware
//
#include <hip/hip_runtime.h>
#include <math.h>

typedef __attribute__((ext_vector_type(16))) __bf16 v16bf;
typedef __attribute__((ext_vector_type(8)))  float  v8f;

#define DEVFN __device__ __forceinline__

static constexpr int Bb   = 8;
static constexpr int HW   = 128 * 128;   // 16384
static constexpr int NT16 = HW / 16;     // 1024 pixel tiles

// ---------------- activation helpers ----------------
template <int ACT> DEVFN float actf(float x) {
  if constexpr (ACT == 1) return fmaxf(x, 0.f);                              // relu
  else if constexpr (ACT == 2) return 0.5f * x * (1.f + erff(x * 0.70710678118654752f)); // exact gelu
  else if constexpr (ACT == 3) return 1.f / (1.f + expf(-x));                // sigmoid
  else return x;
}

DEVFN float wred(float v) {
  #pragma unroll
  for (int o = 16; o; o >>= 1) v += __shfl_xor(v, o, 32);
  return v;
}

// Load one 16x32 A fragment from a row-major [rows x ldk] fp32 matrix whose
// per-lane elements are contiguous: two runs of 8 floats -> 4x float4 loads.
DEVFN v16bf load_a_frag(const float* __restrict__ wrow, int hl) {
  const float4* p0 = (const float4*)(wrow + (hl << 4));
  const float4* p1 = (const float4*)(wrow + 8 + (hl << 4));
  float4 a0 = p0[0], a1 = p0[1], a2 = p1[0], a3 = p1[1];
  v16bf af;
  af[0]  = (__bf16)a0.x; af[1]  = (__bf16)a0.y; af[2]  = (__bf16)a0.z; af[3]  = (__bf16)a0.w;
  af[4]  = (__bf16)a1.x; af[5]  = (__bf16)a1.y; af[6]  = (__bf16)a1.z; af[7]  = (__bf16)a1.w;
  af[8]  = (__bf16)a2.x; af[9]  = (__bf16)a2.y; af[10] = (__bf16)a2.z; af[11] = (__bf16)a2.w;
  af[12] = (__bf16)a3.x; af[13] = (__bf16)a3.y; af[14] = (__bf16)a3.z; af[15] = (__bf16)a3.w;
  return af;
}

// ---------------- pointwise (1x1) conv as WMMA GEMM, O = 64 fixed ----------------
// Y[b,o,n] = act( sum_c W[o,c]*X[b,c,n] (+bias[o]) ) (+Res[b,o,n])
// X has Cin1 channels; channels >= Cin1 come from X2 (channel-concat mode;
// Cin1 % 32 == 0 so each 32-chunk resolves to one source at compile time).
// One wave owns one 16-pixel tile and all four 16-row output tiles:
// B fragment loaded once per K-chunk, 4 preloaded A fragments, 4 back-to-back
// WMMAs (no D->A/B dependency => no hazard NOPs between them).
template <int C, int Cin1, int ACT, bool BIAS, bool RESID>
__global__ void pw_gemm_k(const float* __restrict__ X, const float* __restrict__ X2,
                          const float* __restrict__ Wm, const float* __restrict__ bias,
                          const float* __restrict__ Res, float* __restrict__ Y) {
  int wave = blockIdx.x * (blockDim.x >> 5) + (threadIdx.x >> 5);
  int lane = threadIdx.x & 31;
  int hl = lane >> 4, r = lane & 15;
  int nt = wave % NT16;
  int b  = wave / NT16;
  int n  = nt * 16 + r;

  v8f acc[4] = {};
  #pragma unroll
  for (int k0 = 0; k0 < C; k0 += 32) {
    // B: X[b, k0..k0+31, n] (32x16), loaded once per chunk
    const float* src = (k0 < Cin1)
        ? X  + ((size_t)b * Cin1 + k0) * HW + n
        : X2 + ((size_t)b * Cin1 + (k0 - Cin1)) * HW + n;
    v16bf bf;
    #pragma unroll
    for (int i = 0; i < 16; ++i)
      bf[i] = (__bf16)src[(size_t)((hl << 4) + i) * HW];
    if (k0 + 32 < C && k0 + 32 < Cin1)           // prefetch next K-chunk
      __builtin_prefetch(&X[((size_t)b * Cin1 + k0 + 32 + (hl << 4)) * HW + n], 0, 0);

    v16bf af[4];
    #pragma unroll
    for (int ot = 0; ot < 4; ++ot)
      af[ot] = load_a_frag(Wm + (ot * 16 + r) * C + k0, hl);
    #pragma unroll
    for (int ot = 0; ot < 4; ++ot)
      acc[ot] = __builtin_amdgcn_wmma_f32_16x16x32_bf16(false, af[ot], false, bf,
                                                        (short)0, acc[ot], false, false);
  }
  #pragma unroll
  for (int ot = 0; ot < 4; ++ot) {
    #pragma unroll
    for (int j = 0; j < 8; ++j) {
      int o = ot * 16 + (hl << 3) + j;
      float val = acc[ot][j];
      if (BIAS) val += bias[o];
      val = actf<ACT>(val);
      size_t idx = ((size_t)b * 64 + o) * HW + n;
      if (RESID) val += Res[idx];
      Y[idx] = val;
    }
  }
}

// repack (O=64,C=64,3,3) conv weights -> Wr[tap][o][c] so A fragments are contiguous
__global__ void repack_w9_k(const float* __restrict__ Wm, float* __restrict__ Wr) {
  int i = blockIdx.x * blockDim.x + threadIdx.x;
  if (i >= 9 * 64 * 64) return;
  int kk = i & 63; int o = (i >> 6) & 63; int tap = i >> 12;
  Wr[(tap * 64 + o) * 64 + kk] = Wm[(o * 64 + kk) * 9 + tap];
}

// ---------------- full 3x3 conv (64 -> 64) as 9 shifted WMMA GEMMs ----------------
// Uses repacked weights Wr[tap][o][c]. One wave: one pixel tile, 4 output tiles,
// 72 WMMAs; each shifted 32x16 B tile loaded once, A fragments via b128 loads.
template <int ACT, bool BIAS>
__global__ void conv3x3_gemm_k(const float* __restrict__ X, const float* __restrict__ Wr,
                               const float* __restrict__ bias, float* __restrict__ Y) {
  int wave = blockIdx.x * (blockDim.x >> 5) + (threadIdx.x >> 5);
  int lane = threadIdx.x & 31;
  int hl = lane >> 4, r = lane & 15;
  int nt = wave % NT16;
  int b  = wave / NT16;
  int n  = nt * 16 + r;
  int y  = n >> 7, xx = n & 127;

  v8f acc[4] = {};
  for (int tap = 0; tap < 9; ++tap) {
    int dy = tap / 3 - 1, dx = tap % 3 - 1;
    int sy = y + dy, sx = xx + dx;
    bool ok = ((unsigned)sy < 128u) && ((unsigned)sx < 128u);
    #pragma unroll
    for (int k0 = 0; k0 < 64; k0 += 32) {
      v16bf bf;
      #pragma unroll
      for (int i = 0; i < 16; ++i) {
        int kk = k0 + (hl << 4) + i;
        float v = ok ? X[((size_t)b * 64 + kk) * HW + sy * 128 + sx] : 0.f;
        bf[i] = (__bf16)v;
      }
      v16bf af[4];
      #pragma unroll
      for (int ot = 0; ot < 4; ++ot)
        af[ot] = load_a_frag(Wr + (tap * 64 + ot * 16 + r) * 64 + k0, hl);
      #pragma unroll
      for (int ot = 0; ot < 4; ++ot)
        acc[ot] = __builtin_amdgcn_wmma_f32_16x16x32_bf16(false, af[ot], false, bf,
                                                          (short)0, acc[ot], false, false);
    }
  }
  #pragma unroll
  for (int ot = 0; ot < 4; ++ot) {
    #pragma unroll
    for (int j = 0; j < 8; ++j) {
      int o = ot * 16 + (hl << 3) + j;
      float val = acc[ot][j];
      if (BIAS) val += bias[o];
      val = actf<ACT>(val);
      Y[((size_t)b * 64 + o) * HW + n] = val;
    }
  }
}

// ---------------- depthwise 3x3 conv (channel multiplier O/C) ----------------
template <int ACT, bool BIAS>
__global__ void dwconv2d_k(const float* __restrict__ X, const float* __restrict__ Wm,
                           const float* __restrict__ bias, float* __restrict__ Y,
                           int C, int O) {
  size_t idx = (size_t)blockIdx.x * blockDim.x + threadIdx.x;
  size_t total = (size_t)Bb * O * HW;
  if (idx >= total) return;
  int n = (int)(idx % HW);
  size_t rr = idx / HW;
  int o = (int)(rr % O);
  int b = (int)(rr / O);
  int c = o / (O / C);
  int y = n >> 7, xx = n & 127;
  const float* src = X + ((size_t)b * C + c) * HW;
  float s = 0.f;
  #pragma unroll
  for (int dy = -1; dy <= 1; ++dy) {
    int sy = y + dy; if ((unsigned)sy >= 128u) continue;
    #pragma unroll
    for (int dx = -1; dx <= 1; ++dx) {
      int sx = xx + dx; if ((unsigned)sx >= 128u) continue;
      s += Wm[o * 9 + (dy + 1) * 3 + (dx + 1)] * src[sy * 128 + sx];
    }
  }
  if (BIAS) s += bias[o];
  Y[idx] = actf<ACT>(s);
}

// ---------------- depthwise 3x3x3 conv over (p,h,w), 64 -> 256 ch, GELU ----------------
__global__ void dwconv3d_k(const float* __restrict__ X, const float* __restrict__ Wm,
                           float* __restrict__ Y) {
  size_t idx = (size_t)blockIdx.x * blockDim.x + threadIdx.x;
  size_t total = (size_t)2 * 256 * 4 * HW;
  if (idx >= total) return;
  int n = (int)(idx % HW);
  size_t rr = idx / HW;
  int p  = (int)(rr % 4); rr /= 4;
  int o  = (int)(rr % 256);
  int bn = (int)(rr / 256);
  int c  = o >> 2;
  int y = n >> 7, xx = n & 127;
  float s = 0.f;
  for (int dp = -1; dp <= 1; ++dp) {
    int pp = p + dp; if ((unsigned)pp >= 4u) continue;
    const float* src = X + (((size_t)(bn * 4 + pp)) * 64 + c) * HW;
    #pragma unroll
    for (int dy = -1; dy <= 1; ++dy) {
      int sy = y + dy; if ((unsigned)sy >= 128u) continue;
      #pragma unroll
      for (int dx = -1; dx <= 1; ++dx) {
        int sx = xx + dx; if ((unsigned)sx >= 128u) continue;
        s += Wm[o * 27 + (dp + 1) * 9 + (dy + 1) * 3 + (dx + 1)] * src[sy * 128 + sx];
      }
    }
  }
  Y[(((size_t)(bn * 4 + p)) * 256 + o) * HW + n] = actf<2>(s);
}

// ---------------- LayerNorm over channel dim (64) of NCHW ----------------
__global__ void layernorm_k(const float* __restrict__ X, const float* __restrict__ w,
                            const float* __restrict__ bb, float* __restrict__ Y) {
  size_t idx = (size_t)blockIdx.x * blockDim.x + threadIdx.x;
  if (idx >= (size_t)Bb * HW) return;
  int n = (int)(idx % HW);
  int b = (int)(idx / HW);
  const float* px = X + (size_t)b * 64 * HW + n;
  float* py = Y + (size_t)b * 64 * HW + n;
  float s = 0.f, ss = 0.f;
  for (int c = 0; c < 64; ++c) { float t = px[(size_t)c * HW]; s += t; ss += t * t; }
  float mu  = s * (1.f / 64.f);
  float var = ss * (1.f / 64.f) - mu * mu;
  float inv = rsqrtf(var + 1e-5f);
  for (int c = 0; c < 64; ++c)
    py[(size_t)c * HW] = (px[(size_t)c * HW] - mu) * inv * w[c] + bb[c];
}

// ---------------- spectral attention: gram(QK^T) + row norms, K = 16384 ----------------
__global__ void spec_gram_k(const float* __restrict__ Q, const float* __restrict__ K,
                            float* __restrict__ gram, float* __restrict__ qn,
                            float* __restrict__ kn) {
  int bh = blockIdx.x;                       // b*8 + h, 64 blocks
  const float* qb = Q + (size_t)bh * 8 * HW; // b*64 + h*8 == bh*8
  const float* kb = K + (size_t)bh * 8 * HW;
  float g[64]; float qs[8]; float ks[8];
  #pragma unroll
  for (int t = 0; t < 64; ++t) g[t] = 0.f;
  #pragma unroll
  for (int c = 0; c < 8; ++c) { qs[c] = 0.f; ks[c] = 0.f; }
  for (int n = threadIdx.x; n < HW; n += blockDim.x) {
    float qv[8], kv[8];
    #pragma unroll
    for (int c = 0; c < 8; ++c) { qv[c] = qb[(size_t)c * HW + n]; kv[c] = kb[(size_t)c * HW + n]; }
    #pragma unroll
    for (int c = 0; c < 8; ++c) {
      qs[c] += qv[c] * qv[c]; ks[c] += kv[c] * kv[c];
      #pragma unroll
      for (int d = 0; d < 8; ++d) g[c * 8 + d] += qv[c] * kv[d];
    }
  }
  int lane = threadIdx.x & 31;
  for (int t = 0; t < 64; ++t) {
    float r = wred(g[t]);
    if (lane == 0) atomicAdd(&gram[bh * 64 + t], r);
  }
  for (int c = 0; c < 8; ++c) {
    float r = wred(qs[c]); if (lane == 0) atomicAdd(&qn[bh * 8 + c], r);
    r = wred(ks[c]);       if (lane == 0) atomicAdd(&kn[bh * 8 + c], r);
  }
}

// softmax over d of  gram/(|q||k|)*resc + mb[d] - mb[c]
__global__ void spec_att_k(const float* __restrict__ gram, const float* __restrict__ qn,
                           const float* __restrict__ kn, const float* __restrict__ mb,
                           const float* __restrict__ resc, float* __restrict__ attS) {
  int idx = blockIdx.x * blockDim.x + threadIdx.x;
  if (idx >= 512) return;            // B*8heads*8rows
  int c = idx & 7; int bh = idx >> 3; int h = bh & 7; int b = bh >> 3;
  float qv = fmaxf(sqrtf(qn[bh * 8 + c]), 1e-12f);
  const float* mrow = mb + b * 64 + h * 8;
  float rsc = resc[h];
  float row[8]; float mx = -1e30f;
  #pragma unroll
  for (int d = 0; d < 8; ++d) {
    float kvn = fmaxf(sqrtf(kn[bh * 8 + d]), 1e-12f);
    float a = gram[bh * 64 + c * 8 + d] / (qv * kvn) * rsc + mrow[d] - mrow[c];
    row[d] = a; mx = fmaxf(mx, a);
  }
  float se = 0.f;
  #pragma unroll
  for (int d = 0; d < 8; ++d) { row[d] = expf(row[d] - mx); se += row[d]; }
  float inv = 1.f / se;
  #pragma unroll
  for (int d = 0; d < 8; ++d) attS[bh * 64 + c * 8 + d] = row[d] * inv;
}

// out[b, h*8+c, n] = sum_d att[b,h,c,d] * v[b, h*8+d, n]
__global__ void spec_out_k(const float* __restrict__ attS, const float* __restrict__ V,
                           float* __restrict__ Y) {
  size_t idx = (size_t)blockIdx.x * blockDim.x + threadIdx.x;
  if (idx >= (size_t)Bb * 64 * HW) return;
  int n = (int)(idx % HW);
  size_t rr = idx / HW;
  int cg = (int)(rr % 64);
  int b  = (int)(rr / 64);
  int h = cg >> 3, c = cg & 7;
  const float* vb = V + ((size_t)b * 64 + h * 8) * HW + n;
  const float* at = attS + (b * 8 + h) * 64 + c * 8;
  float s = 0.f;
  #pragma unroll
  for (int d = 0; d < 8; ++d) s += at[d] * vb[(size_t)d * HW];
  Y[idx] = s;
}

// mean over HW per (b, c)
__global__ void meanhw_k(const float* __restrict__ X, float* __restrict__ out) {
  int bc = blockIdx.x;    // 512 blocks
  float s = 0.f;
  for (int n = threadIdx.x; n < HW; n += blockDim.x) s += X[(size_t)bc * HW + n];
  s = wred(s);
  __shared__ float sm[8];
  int lane = threadIdx.x & 31, wv = threadIdx.x >> 5;
  if (lane == 0) sm[wv] = s;
  __syncthreads();
  if (threadIdx.x == 0) {
    float t = 0.f;
    for (int w = 0; w < 8; ++w) t += sm[w];
    out[bc] = t * (1.f / HW);
  }
}

// mean over 64 channels -> mg[bp, n]
__global__ void chmean_k(const float* __restrict__ X, float* __restrict__ mg) {
  size_t idx = (size_t)blockIdx.x * blockDim.x + threadIdx.x;
  if (idx >= (size_t)Bb * HW) return;
  int n = (int)(idx % HW);
  int bp = (int)(idx / HW);
  float s = 0.f;
  for (int c = 0; c < 64; ++c) s += X[((size_t)bp * 64 + c) * HW + n];
  mg[idx] = s * (1.f / 64.f);
}

// ---------------- cross-polar attention core (per pixel, 4x4 over P) ----------------
__global__ void pol_attn_k(const float* __restrict__ Q, const float* __restrict__ K,
                           const float* __restrict__ V, const float* __restrict__ mg,
                           const float* __restrict__ resc, float* __restrict__ Y) {
  size_t idx = (size_t)blockIdx.x * blockDim.x + threadIdx.x;
  if (idx >= (size_t)2 * 8 * HW) return;    // bn * heads * HW
  int n = (int)(idx % HW);
  int rr = (int)(idx / HW);
  int h = rr & 7; int bn = rr >> 3;
  float qv[4][8], kv[4][8], vv[4][8];
  #pragma unroll
  for (int p = 0; p < 4; ++p) {
    const float* bq = Q + (((size_t)(bn * 4 + p)) * 64 + h * 8) * HW + n;
    const float* bk = K + (((size_t)(bn * 4 + p)) * 64 + h * 8) * HW + n;
    const float* bv = V + (((size_t)(bn * 4 + p)) * 64 + h * 8) * HW + n;
    #pragma unroll
    for (int c = 0; c < 8; ++c) {
      qv[p][c] = bq[(size_t)c * HW]; kv[p][c] = bk[(size_t)c * HW]; vv[p][c] = bv[(size_t)c * HW];
    }
  }
  #pragma unroll
  for (int p = 0; p < 4; ++p) {
    float s = 0.f;
    #pragma unroll
    for (int c = 0; c < 8; ++c) s += qv[p][c] * qv[p][c];
    float iq = 1.f / fmaxf(sqrtf(s), 1e-12f);
    s = 0.f;
    #pragma unroll
    for (int c = 0; c < 8; ++c) s += kv[p][c] * kv[p][c];
    float ik = 1.f / fmaxf(sqrtf(s), 1e-12f);
    #pragma unroll
    for (int c = 0; c < 8; ++c) { qv[p][c] *= iq; kv[p][c] *= ik; }
  }
  float m[4];
  #pragma unroll
  for (int p = 0; p < 4; ++p) m[p] = mg[(size_t)(bn * 4 + p) * HW + n];
  float rsc = resc[h];
  float att[4][4];
  #pragma unroll
  for (int p = 0; p < 4; ++p) {
    float row[4]; float mx = -1e30f;
    #pragma unroll
    for (int qq = 0; qq < 4; ++qq) {
      float d = 0.f;
      #pragma unroll
      for (int c = 0; c < 8; ++c) d += qv[p][c] * kv[qq][c];
      row[qq] = rsc * d + m[qq] - m[p];
      mx = fmaxf(mx, row[qq]);
    }
    float se = 0.f;
    #pragma unroll
    for (int qq = 0; qq < 4; ++qq) { row[qq] = expf(row[qq] - mx); se += row[qq]; }
    float inv = 1.f / se;
    #pragma unroll
    for (int qq = 0; qq < 4; ++qq) att[p][qq] = row[qq] * inv;
  }
  #pragma unroll
  for (int p = 0; p < 4; ++p)
    #pragma unroll
    for (int c = 0; c < 8; ++c) {
      float o = 0.f;
      #pragma unroll
      for (int qq = 0; qq < 4; ++qq) o += att[p][qq] * vv[qq][c];
      Y[(((size_t)(bn * 4 + p)) * 64 + h * 8 + c) * HW + n] = o;
    }
}

// ---------------- misc elementwise ----------------
__global__ void copy2_k(const float* __restrict__ X, float* __restrict__ xs,
                        float* __restrict__ xp) {
  size_t idx = (size_t)blockIdx.x * blockDim.x + threadIdx.x;
  if (idx >= (size_t)Bb * 64 * HW) return;
  float v = X[idx]; xs[idx] = v; xp[idx] = v;
}

__global__ void blend_k(const float* __restrict__ A, const float* __restrict__ XS,
                        const float* __restrict__ XP, float* __restrict__ out) {
  size_t idx = (size_t)blockIdx.x * blockDim.x + threadIdx.x;
  if (idx >= (size_t)Bb * 64 * HW) return;
  float a = A[idx];
  out[idx] = a * XS[idx] + (1.f - a) * XP[idx];
}

__global__ void zero_k(float* __restrict__ p, int n) {
  int i = blockIdx.x * blockDim.x + threadIdx.x;
  if (i < n) p[i] = 0.f;
}

// ---------------- orchestration ----------------
extern "C" void kernel_launch(void* const* d_in, const int* in_sizes, int n_in,
                              void* d_out, int out_size, void* d_ws, size_t ws_size,
                              hipStream_t stream) {
  (void)in_sizes; (void)n_in; (void)out_size; (void)ws_size;
  const float* x     = (const float*)d_in[0];
  const float* mask  = (const float*)d_in[1];
  const float* sl1w  = (const float*)d_in[2];
  const float* sl1b  = (const float*)d_in[3];
  const float* sl2w  = (const float*)d_in[4];
  const float* sl2b  = (const float*)d_in[5];
  const float* swq   = (const float*)d_in[6];
  const float* swk   = (const float*)d_in[7];
  const float* swv   = (const float*)d_in[8];
  const float* sproj = (const float*)d_in[9];
  const float* smp2  = (const float*)d_in[10];
  const float* sdq   = (const float*)d_in[11];
  const float* sdk   = (const float*)d_in[12];
  const float* sdv   = (const float*)d_in[13];
  const float* smp1  = (const float*)d_in[14];
  const float* sresc = (const float*)d_in[15];
  const float* sf1   = (const float*)d_in[16];
  const float* sf2   = (const float*)d_in[17];
  const float* pl1w  = (const float*)d_in[18];
  const float* pl1b  = (const float*)d_in[19];
  const float* pl2w  = (const float*)d_in[20];
  const float* pl2b  = (const float*)d_in[21];
  const float* pwq   = (const float*)d_in[22];
  const float* pwk   = (const float*)d_in[23];
  const float* pwv   = (const float*)d_in[24];
  const float* ppw   = (const float*)d_in[25];
  const float* pm2w  = (const float*)d_in[26];
  const float* ppb   = (const float*)d_in[27];
  const float* presc = (const float*)d_in[28];
  const float* pm1w  = (const float*)d_in[29];
  const float* pm1b  = (const float*)d_in[30];
  const float* pm2b  = (const float*)d_in[31];
  const float* pf1   = (const float*)d_in[32];
  const float* pf2   = (const float*)d_in[33];
  const float* fw1   = (const float*)d_in[34];
  const float* fb1   = (const float*)d_in[35];
  const float* fw2   = (const float*)d_in[36];
  const float* fb2   = (const float*)d_in[37];
  float* out = (float*)d_out;

  float* ws = (float*)d_ws;
  const size_t T = (size_t)Bb * 64 * HW;        // 8,388,608 floats per tensor
  float* xs   = ws + 0 * T;
  float* xp   = ws + 1 * T;
  float* ln   = ws + 2 * T;
  float* q    = ws + 3 * T;
  float* k    = ws + 4 * T;
  float* v    = ws + 5 * T;
  float* t1   = ws + 6 * T;
  float* t2   = ws + 7 * T;
  float* t256 = ws + 8 * T;                     // 4T (256-ch temp)
  float* smem = ws + 12 * T;                    // small scratch
  float* gram = smem;                           // 4096
  float* qn   = smem + 4096;                    // 512
  float* kn   = smem + 4608;                    // 512
  float* mb   = smem + 6144;                    // 512
  float* attS = smem + 8192;                    // 4096
  float* mg   = smem + 16384;                   // 131072
  float* wrep = smem + 16384 + 131072;          // 36864 (repacked 3x3 weights)

  const int TPB = 256;
  auto g1 = [](size_t n) { return dim3((unsigned)((n + 255) / 256)); };
  dim3 gPW((unsigned)(Bb * NT16 / 8));          // 1024 blocks, 8 waves each: one wave per (b, pixel-tile)

  copy2_k<<<g1(T), TPB, 0, stream>>>(x, xs, xp);

  // ================= spectral branch (2 layers) =================
  for (int i = 0; i < 2; ++i) {
    layernorm_k<<<g1((size_t)Bb * HW), TPB, 0, stream>>>(xs, sl1w + i * 64, sl1b + i * 64, ln);
    pw_gemm_k<64, 64, 0, false, false><<<gPW, TPB, 0, stream>>>(ln, nullptr, swq + i * 4096, nullptr, nullptr, t1);
    dwconv2d_k<0, false><<<g1(T), TPB, 0, stream>>>(t1, sdq + i * 576, nullptr, q, 64, 64);
    pw_gemm_k<64, 64, 0, false, false><<<gPW, TPB, 0, stream>>>(ln, nullptr, swk + i * 4096, nullptr, nullptr, t1);
    dwconv2d_k<0, false><<<g1(T), TPB, 0, stream>>>(t1, sdk + i * 576, nullptr, k, 64, 64);
    pw_gemm_k<64, 64, 0, false, false><<<gPW, TPB, 0, stream>>>(ln, nullptr, swv + i * 4096, nullptr, nullptr, t1);
    dwconv2d_k<0, false><<<g1(T), TPB, 0, stream>>>(t1, sdv + i * 576, nullptr, v, 64, 64);

    zero_k<<<g1(5120), TPB, 0, stream>>>(smem, 5120);                  // gram + qn + kn
    spec_gram_k<<<64, TPB, 0, stream>>>(q, k, gram, qn, kn);

    repack_w9_k<<<g1(36864), TPB, 0, stream>>>(smp1 + i * 36864, wrep);
    conv3x3_gemm_k<1, false><<<gPW, TPB, 0, stream>>>(mask, wrep, nullptr, t1); // relu fused
    pw_gemm_k<64, 64, 0, false, false><<<gPW, TPB, 0, stream>>>(t1, nullptr, smp2 + i * 4096, nullptr, nullptr, t2);
    meanhw_k<<<512, TPB, 0, stream>>>(t2, mb);

    spec_att_k<<<2, TPB, 0, stream>>>(gram, qn, kn, mb, sresc + i * 8, attS);
    spec_out_k<<<g1(T), TPB, 0, stream>>>(attS, v, t1);
    pw_gemm_k<64, 64, 0, false, true><<<gPW, TPB, 0, stream>>>(t1, nullptr, sproj + i * 4096, nullptr, xs, xs);

    layernorm_k<<<g1((size_t)Bb * HW), TPB, 0, stream>>>(xs, sl2w + i * 64, sl2b + i * 64, ln);
    dwconv2d_k<2, false><<<g1(4 * T), TPB, 0, stream>>>(ln, sf1 + i * 2304, nullptr, t256, 64, 256); // gelu
    pw_gemm_k<256, 256, 0, false, true><<<gPW, TPB, 0, stream>>>(t256, nullptr, sf2 + i * 16384, nullptr, xs, xs);
  }

  // ================= cross-polar branch (2 layers) =================
  for (int i = 0; i < 2; ++i) {
    layernorm_k<<<g1((size_t)Bb * HW), TPB, 0, stream>>>(xp, pl1w + i * 64, pl1b + i * 64, ln);
    pw_gemm_k<64, 64, 0, false, false><<<gPW, TPB, 0, stream>>>(ln, nullptr, pwq + i * 4096, nullptr, nullptr, q);
    pw_gemm_k<64, 64, 0, false, false><<<gPW, TPB, 0, stream>>>(ln, nullptr, pwk + i * 4096, nullptr, nullptr, k);
    pw_gemm_k<64, 64, 0, false, false><<<gPW, TPB, 0, stream>>>(ln, nullptr, pwv + i * 4096, nullptr, nullptr, v);

    dwconv2d_k<2, true><<<g1(T), TPB, 0, stream>>>(mask, pm1w + i * 576, pm1b + i * 64, t1, 64, 64); // bias+gelu
    pw_gemm_k<64, 64, 0, true, false><<<gPW, TPB, 0, stream>>>(t1, nullptr, pm2w + i * 4096, pm2b + i * 64, nullptr, t2);
    chmean_k<<<g1((size_t)Bb * HW), TPB, 0, stream>>>(t2, mg);

    pol_attn_k<<<g1((size_t)2 * 8 * HW), TPB, 0, stream>>>(q, k, v, mg, presc + i * 8, t1);
    pw_gemm_k<64, 64, 0, true, true><<<gPW, TPB, 0, stream>>>(t1, nullptr, ppw + i * 4096, ppb + i * 64, xp, xp);

    layernorm_k<<<g1((size_t)Bb * HW), TPB, 0, stream>>>(xp, pl2w + i * 64, pl2b + i * 64, ln);
    dwconv3d_k<<<g1(4 * T), TPB, 0, stream>>>(ln, pf1 + i * 6912, t256);  // gelu fused
    pw_gemm_k<256, 256, 0, false, true><<<gPW, TPB, 0, stream>>>(t256, nullptr, pf2 + i * 16384, nullptr, xp, xp);
  }

  // ================= sigmoid fusion =================
  pw_gemm_k<128, 64, 1, true, false><<<gPW, TPB, 0, stream>>>(xs, xp, fw1, fb1, nullptr, t1); // concat + bias + relu
  repack_w9_k<<<g1(36864), TPB, 0, stream>>>(fw2, wrep);
  conv3x3_gemm_k<3, true><<<gPW, TPB, 0, stream>>>(t1, wrep, fb2, t2);                        // bias + sigmoid
  blend_k<<<g1(T), TPB, 0, stream>>>(t2, xs, xp, out);
}
